// BDHGraphModel_36636071035465
// MI455X (gfx1250) — matrix operand order repaired and checked by reading, not compiled, not run on hardware
//
#include <hip/hip_runtime.h>
#include <hip/hip_bf16.h>
#include <stdint.h>

// ---------------------------------------------------------------------------
// Problem constants (from the reference)
// ---------------------------------------------------------------------------
#define N_NEURONS 4096
#define N_EDGES   131072
#define N_LAYERS  2
#define VOCAB     8192
#define BB        8      // batch
#define TT        128    // time steps
#define DECAY     0.99f
#define NB        (N_NEURONS * BB)   // 32768 node-state elements (N, B) layout

#define NWG_RECUR     128
#define RECUR_THREADS 256            // 128*256 == NB exactly: thread <-> (node d, batch b)

typedef __bf16 v16bf __attribute__((ext_vector_type(16)));
typedef float  v8f   __attribute__((ext_vector_type(8)));

__device__ __forceinline__ unsigned short f2bf(float f) {
  // round-to-nearest-even f32 -> bf16
  unsigned u = __float_as_uint(f);
  u += 0x7FFFu + ((u >> 16) & 1u);
  return (unsigned short)(u >> 16);
}

// ---------------------------------------------------------------------------
// 1) Embedding gather: Xin[t][n][b] = Wemb[idx[b][t]][n]   (layout (T, N, B))
// ---------------------------------------------------------------------------
__global__ void k_embed(const float* __restrict__ Wemb, const int* __restrict__ idx,
                        float* __restrict__ Xin) {
  const int t = blockIdx.x;
  __shared__ int rowbase[BB];
  if (threadIdx.x < BB) rowbase[threadIdx.x] = idx[threadIdx.x * TT + t] * N_NEURONS;
  __syncthreads();
  float* outt = Xin + (size_t)t * NB;
  for (int n = threadIdx.x; n < N_NEURONS; n += blockDim.x) {
    float v[BB];
#pragma unroll
    for (int b = 0; b < BB; ++b) v[b] = Wemb[(size_t)rowbase[b] + n];
    float4* o = (float4*)(outt + (size_t)n * BB);
    o[0] = make_float4(v[0], v[1], v[2], v[3]);
    o[1] = make_float4(v[4], v[5], v[6], v[7]);
  }
}

// ---------------------------------------------------------------------------
// 2) CSR-by-dst build (turns all scatters into gathers -> no hot-loop atomics)
// ---------------------------------------------------------------------------
__global__ void k_count(const int* __restrict__ dst, int* counts) {
  const int e = blockIdx.x * blockDim.x + threadIdx.x;
  if (e < N_EDGES) atomicAdd(&counts[dst[e]], 1);
}

__global__ void k_scan(const int* __restrict__ counts, int* __restrict__ offsets,
                       int* __restrict__ cursor) {
  // 256 threads, 16 elements each, exclusive scan of 4096 counts
  __shared__ int part[256];
  const int tid = threadIdx.x;
  const int base = tid * 16;
  int loc[16];
  int s = 0;
#pragma unroll
  for (int i = 0; i < 16; ++i) { loc[i] = s; s += counts[base + i]; }
  part[tid] = s;
  __syncthreads();
  for (int off = 1; off < 256; off <<= 1) {
    int v = (tid >= off) ? part[tid - off] : 0;
    __syncthreads();
    part[tid] += v;
    __syncthreads();
  }
  const int prev = (tid == 0) ? 0 : part[tid - 1];
#pragma unroll
  for (int i = 0; i < 16; ++i) {
    const int o = prev + loc[i];
    offsets[base + i] = o;
    cursor[base + i]  = o;
  }
  if (tid == 255) offsets[N_NEURONS] = part[255];
}

__global__ void k_fill(const int* __restrict__ dst, int* cursor, int* __restrict__ csr_e) {
  const int e = blockIdx.x * blockDim.x + threadIdx.x;
  if (e < N_EDGES) {
    const int p = atomicAdd(&cursor[dst[e]], 1);
    csr_e[p] = e;
  }
}

// ---------------------------------------------------------------------------
// Grid-wide barrier (monotonic arrival count, s_sleep spin)
// ---------------------------------------------------------------------------
__device__ __forceinline__ void grid_barrier(unsigned* cnt, unsigned& gen) {
  __threadfence();
  __syncthreads();
  ++gen;
  if (threadIdx.x == 0) {
    __hip_atomic_fetch_add(cnt, 1u, __ATOMIC_ACQ_REL, __HIP_MEMORY_SCOPE_AGENT);
    const unsigned target = gen * (unsigned)NWG_RECUR;
    while (__hip_atomic_load(cnt, __ATOMIC_ACQUIRE, __HIP_MEMORY_SCOPE_AGENT) < target) {
      __builtin_amdgcn_s_sleep(2);
    }
  }
  __syncthreads();
  __threadfence();
}

// ---------------------------------------------------------------------------
// 3) Persistent recurrence kernel. Node state layout (N, B): elem = d*8 + b.
//    Thread gid owns (d = gid>>3, b = gid&7); the 8 batch lanes of a node are
//    8 adjacent lanes of a wave (cross-b hebbian reduce via __shfl_xor).
//    xs is emitted as bf16 with relu applied (feeds WMMA GEMM directly).
//    sigma points into the tail of d_out (zero-initialized here).
// ---------------------------------------------------------------------------
__global__ void __launch_bounds__(RECUR_THREADS) k_recur(
    const float* __restrict__ Xin, const int* __restrict__ src,
    const float* __restrict__ Gx, const float* __restrict__ Gy,
    const float* __restrict__ Gs,
    const int* __restrict__ csr_off, const int* __restrict__ csr_e,
    float* __restrict__ sigma, unsigned short* __restrict__ xs_bf,
    float* xcur, float* y0, float* y1, float* Abuf, unsigned* barcnt) {
  const int gid = blockIdx.x * RECUR_THREADS + threadIdx.x;  // 0..NB-1
  const int b   = gid & 7;
  const int d   = gid >> 3;
  unsigned gen = 0;

  // pre-phase: sigma = 0, y0 = X[:,0,:]
  for (int e = gid; e < N_EDGES; e += NB) sigma[e] = 0.0f;
  y0[gid] = Xin[gid];
  grid_barrier(barcnt, gen);

  const int e0 = csr_off[d];
  const int e1 = csr_off[d + 1];
  float* yold = y0;
  float* ynew = y1;

  for (int t = 0; t < TT; ++t) {
    const float* xin_t = Xin + (size_t)t * NB;
    for (int l = 0; l < N_LAYERS; ++l) {
      // ---- P1: A[d,b] = sum_e x[src,b]*sigma_old; hebbian + sigma update ----
      const float xd = (l == 0) ? xin_t[gid] : fmaxf(xcur[gid], 0.0f);
      float acc = 0.0f;
      for (int p = e0; p < e1; ++p) {
        const int   e  = csr_e[p];
        const int   s  = src[e];
        const float sg = sigma[e];  // read by all 8 lanes before lane b==0 updates
        const float xs_v = (l == 0) ? xin_t[s * BB + b] : fmaxf(xcur[s * BB + b], 0.0f);
        float h = yold[s * BB + b] * xd;
        h += __shfl_xor(h, 1);
        h += __shfl_xor(h, 2);
        h += __shfl_xor(h, 4);                       // sum over 8 batch lanes
        if (b == 0) sigma[e] = (sg + h * 0.125f * Gs[e]) * DECAY;
        acc += xs_v * sg;                            // uses sigma before update
      }
      Abuf[gid] = acc;
      grid_barrier(barcnt, gen);

      // ---- P2: y_new[d,b] = sum_e relu(A[src,b]) * Gy[e] ----
      acc = 0.0f;
      for (int p = e0; p < e1; ++p) {
        const int e = csr_e[p];
        const int s = src[e];
        acc += fmaxf(Abuf[s * BB + b], 0.0f) * Gy[e];
      }
      ynew[gid] = acc;
      grid_barrier(barcnt, gen);

      // ---- P3: x[d,b] = sum_e y_new[src,b] * Gx[e]  (relu applied on read) ----
      acc = 0.0f;
      for (int p = e0; p < e1; ++p) {
        const int e = csr_e[p];
        const int s = src[e];
        acc += ynew[s * BB + b] * Gx[e];
      }
      xcur[gid] = acc;
      if (l == N_LAYERS - 1)
        xs_bf[(size_t)t * NB + gid] = f2bf(fmaxf(acc, 0.0f));
      grid_barrier(barcnt, gen);

      float* tmp = yold; yold = ynew; ynew = tmp;
    }
  }
}

// ---------------------------------------------------------------------------
// 4) logits = einsum('tbn,vn->btv', xs, W_out) + b_out  via bf16 WMMA.
//    A = xs (M=1024 rows r=t*8+b, K=4096), B = W_out[v][k] (K x N=8192).
//    Block tile 64(M) x 256(N); 8 waves: wave = (m_half 0..1, colgrp 0..3);
//    each wave: 2 m-subtiles x 4 n-subtiles of v_wmma_f32_16x16x32_bf16.
// ---------------------------------------------------------------------------
#define GTM 64
#define GTN 256
#define GTK 32

__global__ void __launch_bounds__(256) k_logits(
    const unsigned short* __restrict__ xs_bf, const float* __restrict__ W_out,
    const float* __restrict__ b_out, float* __restrict__ out) {
  __shared__ __align__(16) unsigned short As[GTM][GTK];   // 4 KB
  __shared__ __align__(16) unsigned short Bs[GTN][GTK];   // 16 KB

  const int tid    = threadIdx.x;
  const int wave   = tid >> 5;
  const int lane   = tid & 31;
  const int lrow   = lane & 15;
  const bool hi    = lane >= 16;
  const int m_half = wave >> 2;   // 0..1
  const int cgrp   = wave & 3;    // 0..3
  const int block_m = blockIdx.x * GTM;      // grid.x = 1024/64 = 16
  const int block_n = blockIdx.y * GTN;      // grid.y = 8192/256 = 32
  const int t0      = block_m >> 3;          // 8 time steps per block row-tile

  const v8f zero = {0.f, 0.f, 0.f, 0.f, 0.f, 0.f, 0.f, 0.f};
  v8f acc[2][4];
#pragma unroll
  for (int mi = 0; mi < 2; ++mi)
#pragma unroll
    for (int c = 0; c < 4; ++c) acc[mi][c] = zero;

  union Frag { v16bf v; uint4 u[2]; };

  for (int kk = 0; kk < N_NEURONS; kk += GTK) {
    // --- stage A tile: source xs_bf[t][k][b], dest As[(t-t0)*8+b][k] ---
    {
      const int i   = tid * 8;            // 8 consecutive bf16 = one uint4
      const int c   = i >> 8;             // chunk (time step offset)
      const int k   = (i & 255) >> 3;     // column within tile
      const uint4 vsrc = *(const uint4*)(xs_bf + (size_t)(t0 + c) * NB +
                                         (size_t)(kk + k) * BB);
      union { uint4 u; unsigned short s[8]; } uu;
      uu.u = vsrc;
#pragma unroll
      for (int bb2 = 0; bb2 < 8; ++bb2) As[c * 8 + bb2][k] = uu.s[bb2];
    }
    // --- stage B tile: W_out rows (block_n..+256), k chunk, f32 -> bf16 ---
#pragma unroll
    for (int p = 0; p < 8; ++p) {
      const int n  = p * 32 + (tid >> 3);
      const int k4 = (tid & 7) * 4;
      const float4 w = *(const float4*)(W_out + (size_t)(block_n + n) * N_NEURONS + kk + k4);
      unsigned short* bp = &Bs[n][k4];
      bp[0] = f2bf(w.x); bp[1] = f2bf(w.y); bp[2] = f2bf(w.z); bp[3] = f2bf(w.w);
    }
    __syncthreads();

    // --- per-lane fragment gather, contiguous k order (k = 16h + s) for A and B ---
    // A 16x32: lanes 0-15: row=M, K 0..15; lanes 16-31: K 16..31
    Frag af[2];
#pragma unroll
    for (int mi = 0; mi < 2; ++mi) {
      const int row = m_half * 32 + mi * 16 + lrow;
      const uint4* ap = (const uint4*)&As[row][hi ? 16 : 0];
      af[mi].u[0] = ap[0];
      af[mi].u[1] = ap[1];
    }
    // B 32x16: lanes 0-15: col=N, K 0..15; lanes 16-31: K 16..31
    Frag bfr[4];
#pragma unroll
    for (int c = 0; c < 4; ++c) {
      const int n = cgrp * 64 + c * 16 + lrow;
      const uint4* bp = (const uint4*)&Bs[n][hi ? 16 : 0];
      bfr[c].u[0] = bp[0];
      bfr[c].u[1] = bp[1];
    }
#pragma unroll
    for (int mi = 0; mi < 2; ++mi)
#pragma unroll
      for (int c = 0; c < 4; ++c)
        acc[mi][c] = __builtin_amdgcn_wmma_f32_16x16x32_bf16(
            false, af[mi].v, false, bfr[c].v, (short)0, acc[mi][c], false, false);
    __syncthreads();
  }

  // --- epilogue: C/D layout -> logits (B,T,V), add b_out ---
#pragma unroll
  for (int c = 0; c < 4; ++c) {
    const int n  = block_n + cgrp * 64 + c * 16 + lrow;
    const float bo = b_out[n];
#pragma unroll
    for (int mi = 0; mi < 2; ++mi) {
#pragma unroll
      for (int v = 0; v < 8; ++v) {
        const int m    = block_m + m_half * 32 + mi * 16 + v + (hi ? 8 : 0);
        const int bidx = m & 7;        // r = t*8 + b
        const int tt   = m >> 3;
        out[(size_t)bidx * (TT * VOCAB) + (size_t)tt * VOCAB + n] = acc[mi][c][v] + bo;
      }
    }
  }
}

// ---------------------------------------------------------------------------
// Host launcher
// ---------------------------------------------------------------------------
extern "C" void kernel_launch(void* const* d_in, const int* in_sizes, int n_in,
                              void* d_out, int out_size, void* d_ws, size_t ws_size,
                              hipStream_t stream) {
  (void)in_sizes; (void)n_in; (void)out_size; (void)ws_size;

  const int*   idx   = (const int*)  d_in[0];
  const int*   src   = (const int*)  d_in[1];
  const int*   dst   = (const int*)  d_in[2];
  const float* Wemb  = (const float*)d_in[3];
  const float* Gx    = (const float*)d_in[4];
  const float* Gy    = (const float*)d_in[5];
  const float* Gs    = (const float*)d_in[6];
  const float* W_out = (const float*)d_in[7];
  const float* b_out = (const float*)d_in[8];

  float* logits = (float*)d_out;
  float* sigma  = logits + (size_t)BB * TT * VOCAB;   // sigma_final lives in d_out tail

  char* ws = (char*)d_ws;
  size_t off = 0;
  auto take = [&](size_t bytes) -> char* {
    char* p = ws + off;
    off = (off + bytes + 255) & ~(size_t)255;
    return p;
  };
  float*          Xin     = (float*)          take((size_t)TT * NB * sizeof(float));          // 16 MB
  unsigned short* xs_bf   = (unsigned short*) take((size_t)TT * NB * sizeof(unsigned short)); // 8 MB
  float*          xcur    = (float*)          take(NB * sizeof(float));
  float*          ybuf0   = (float*)          take(NB * sizeof(float));
  float*          ybuf1   = (float*)          take(NB * sizeof(float));
  float*          Abuf    = (float*)          take(NB * sizeof(float));
  int*            csr_e   = (int*)            take(N_EDGES * sizeof(int));
  int*            csr_off = (int*)            take((N_NEURONS + 1) * sizeof(int));
  char*           zbase   =                   ws + off;
  int*            counts  = (int*)            take(N_NEURONS * sizeof(int));
  int*            cursor  = (int*)            take(N_NEURONS * sizeof(int));
  unsigned*       barcnt  = (unsigned*)       take(256);
  const size_t zbytes = (size_t)(((char*)barcnt + 256) - zbase);

  hipMemsetAsync(zbase, 0, zbytes, stream);  // counts, cursor, barrier counter

  hipLaunchKernelGGL(k_embed, dim3(TT), dim3(256), 0, stream, Wemb, idx, Xin);
  hipLaunchKernelGGL(k_count, dim3(N_EDGES / 256), dim3(256), 0, stream, dst, counts);
  hipLaunchKernelGGL(k_scan,  dim3(1), dim3(256), 0, stream, counts, csr_off, cursor);
  hipLaunchKernelGGL(k_fill,  dim3(N_EDGES / 256), dim3(256), 0, stream, dst, cursor, csr_e);
  hipLaunchKernelGGL(k_recur, dim3(NWG_RECUR), dim3(RECUR_THREADS), 0, stream,
                     Xin, src, Gx, Gy, Gs, csr_off, csr_e, sigma, xs_bf,
                     xcur, ybuf0, ybuf1, Abuf, barcnt);
  hipLaunchKernelGGL(k_logits, dim3(16, 32), dim3(256), 0, stream,
                     xs_bf, W_out, b_out, logits);
}
